// RNNMany2One_26551487823986
// MI455X (gfx1250) — matrix operand order repaired and checked by reading, not compiled, not run on hardware
//
#include <hip/hip_runtime.h>
#include <math.h>

#define NB   128
#define LSEQ 1024
#define IN   256
#define H    512

typedef __bf16 v16bf __attribute__((ext_vector_type(16)));
typedef float  v8f   __attribute__((ext_vector_type(8)));
typedef unsigned short usht8 __attribute__((ext_vector_type(8)));

union Frag16 { v16bf v; usht8 u8[2]; };

__device__ __forceinline__ unsigned short f2bf(float f) {
    unsigned u = __builtin_bit_cast(unsigned, f);
    u += 0x7FFFu + ((u >> 16) & 1u);          // round-to-nearest-even
    return (unsigned short)(u >> 16);
}
__device__ __forceinline__ float bf2f(unsigned short b) {
    unsigned u = ((unsigned)b) << 16;
    return __builtin_bit_cast(float, u);
}

// ---------------------------------------------------------------------------
// Kernel A: transpose + bf16-convert W_hh and W_out into [K][N] layout so the
// scan kernel's WMMA B-fragments are single contiguous 32B loads (L2-resident).
// ---------------------------------------------------------------------------
__global__ __launch_bounds__(512) void wprep_kernel(
    const float* __restrict__ whh, const float* __restrict__ wout,
    unsigned short* __restrict__ whhT, unsigned short* __restrict__ woutT) {
    int idx = blockIdx.x * blockDim.x + threadIdx.x;   // 0 .. 262143
    int k = idx >> 9, n = idx & 511;
    whhT [idx] = f2bf(whh [(size_t)n * H + k]);
    woutT[idx] = f2bf(wout[(size_t)n * H + k]);
}

// ---------------------------------------------------------------------------
// xp workspace layout, pre-swizzled into WMMA C-fragment order for the scan:
//   element (batch nb, time t, col ncol):
//     bt = nb>>4, r = nb&15, lh = r>>3, v = r&7
//     addr = ((((bt*LSEQ + t)*2 + lh)*H + ncol)*8 + v
// so a scan lane (lh, lm) reads its 8 C values for a tile as ONE 16B load.
// ---------------------------------------------------------------------------
__device__ __forceinline__ size_t xp_addr(int bt, int t, int lh, int ncol) {
    return ((((size_t)bt * LSEQ + t) * 2 + lh) * H + ncol) * 8;
}

// ---------------------------------------------------------------------------
// Kernel B: xp[m,h] = sum_i x[m,i]*W_ih[h,i] + b_ih[h] + b_hh[h], m = nb*L + t.
// 128x128 output tile per workgroup, K staged through LDS in chunks of 64.
// 8 waves; wave w computes rows [w*16, w*16+16) x all 128 cols (8 WMMA tiles).
// ---------------------------------------------------------------------------
__global__ __launch_bounds__(256) void xp_gemm_kernel(
    const float* __restrict__ x, const float* __restrict__ w_ih,
    const float* __restrict__ b_ih, const float* __restrict__ b_hh,
    unsigned short* __restrict__ xp) {
    __shared__ unsigned short As[128][72];    // M x Kc(64), +8 pad, 16B-aligned rows
    __shared__ unsigned short Bs[64][136];    // Kc x N(128), +8 pad

    const int m0 = blockIdx.x * 128;
    const int h0 = blockIdx.y * 128;
    const int tid  = threadIdx.x;
    const int wave = tid >> 5, lane = tid & 31;
    const int lh = lane >> 4, lm = lane & 15;   // half, lane%16

    v8f acc[8];
    #pragma unroll
    for (int t = 0; t < 8; ++t)
        #pragma unroll
        for (int e = 0; e < 8; ++e) acc[t][e] = 0.0f;

    for (int ko = 0; ko < 4; ++ko) {
        // x chunk -> As  (128 rows x 64 cols fp32 -> bf16)
        #pragma unroll
        for (int j = 0; j < 8; ++j) {
            int idx = tid * 8 + j;
            int r = idx >> 4, c4 = idx & 15;
            float4 xv = *(const float4*)(x + (size_t)(m0 + r) * IN + ko * 64 + c4 * 4);
            As[r][c4 * 4 + 0] = f2bf(xv.x);
            As[r][c4 * 4 + 1] = f2bf(xv.y);
            As[r][c4 * 4 + 2] = f2bf(xv.z);
            As[r][c4 * 4 + 3] = f2bf(xv.w);
        }
        // W_ih chunk -> Bs transposed: Bs[k][n] = W_ih[h0+n][k]
        #pragma unroll
        for (int j = 0; j < 8; ++j) {
            int idx = tid * 8 + j;
            int r = idx >> 4, c4 = idx & 15;    // r = n, c4*4 = k-local
            float4 wv = *(const float4*)(w_ih + (size_t)(h0 + r) * IN + ko * 64 + c4 * 4);
            Bs[c4 * 4 + 0][r] = f2bf(wv.x);
            Bs[c4 * 4 + 1][r] = f2bf(wv.y);
            Bs[c4 * 4 + 2][r] = f2bf(wv.z);
            Bs[c4 * 4 + 3][r] = f2bf(wv.w);
        }
        __syncthreads();
        #pragma unroll
        for (int kc = 0; kc < 2; ++kc) {
            Frag16 a;
            const int ar = wave * 16 + lm;
            const int kb = kc * 32 + lh * 8;
            a.u8[0] = *(const usht8*)&As[ar][kb];
            a.u8[1] = *(const usht8*)&As[ar][kb + 16];
            const int kr = kc * 32 + lh * 8;
            #pragma unroll
            for (int nt = 0; nt < 8; ++nt) {
                Frag16 b;
                #pragma unroll
                for (int i = 0; i < 8; ++i) {
                    b.u8[0][i] = Bs[kr + i][nt * 16 + lm];
                    b.u8[1][i] = Bs[kr + 16 + i][nt * 16 + lm];
                }
                acc[nt] = __builtin_amdgcn_wmma_f32_16x16x32_bf16(
                    false, a.v, false, b.v, (short)0, acc[nt], false, false);
            }
        }
        __syncthreads();
    }
    // bias + store bf16 xp in scan-fragment layout
    #pragma unroll
    for (int nt = 0; nt < 8; ++nt) {
        const int ncol = h0 + nt * 16 + lm;
        const float bias = b_ih[ncol] + b_hh[ncol];
        #pragma unroll
        for (int v = 0; v < 8; ++v) {
            const int m  = m0 + wave * 16 + lh * 8 + v;   // flat row = nb*L + t
            const int nb = m >> 10;                       // L == 1024
            const int tt = m & (LSEQ - 1);
            const int r  = nb & 15;
            xp[xp_addr(nb >> 4, tt, r >> 3, ncol) + (r & 7)] =
                f2bf(acc[nt][v] + bias);
        }
    }
}

// ---------------------------------------------------------------------------
// Kernel C: the recurrent scan + fused output projection.
// 8 workgroups, each owns a 16-row batch tile (independent recurrences).
// 512 threads = 16 waves; wave w owns output cols [w*32, w*32+32) (2 tiles).
// h double-buffered in LDS (bf16) -> ONE barrier per step; W_hh^T streams
// from L2; xp fragments software-pipelined one step ahead + prefetch 2 ahead.
// ---------------------------------------------------------------------------
__global__ __launch_bounds__(512) void rnn_scan_kernel(
    const unsigned short* __restrict__ xp,
    const float* __restrict__ h0in,
    const unsigned short* __restrict__ whhT,
    const unsigned short* __restrict__ woutT,
    const float* __restrict__ b_out,
    float* __restrict__ y_out,
    float* __restrict__ h_out) {
    __shared__ unsigned short hs[2][16][H + 8];   // row stride 520*2B, 16B aligned

    const int bt   = blockIdx.x;            // batch tile
    const int b0   = bt * 16;
    const int tid  = threadIdx.x;
    const int wave = tid >> 5, lane = tid & 31;
    const int lh = lane >> 4, lm = lane & 15;
    const int n0 = wave * 32;

    // init h tile from input h (fp32 -> bf16) into buffer 0
    for (int i = tid; i < 16 * H; i += 512) {
        int r = i >> 9, c = i & 511;
        hs[0][r][c] = f2bf(h0in[(size_t)(b0 + r) * H + c]);
    }

    const int nc0 = n0 + lm, nc1 = n0 + 16 + lm;
    usht8 xf0 = *(const usht8*)(xp + xp_addr(bt, 0, lh, nc0));
    usht8 xf1 = *(const usht8*)(xp + xp_addr(bt, 0, lh, nc1));
    __syncthreads();

    for (int t = 0; t < LSEQ; ++t) {
        const int cur = t & 1, nxt = cur ^ 1;

        v8f acc[2];
        #pragma unroll
        for (int v = 0; v < 8; ++v) { acc[0][v] = bf2f(xf0[v]); acc[1][v] = bf2f(xf1[v]); }

        // software pipeline: fetch xp for t+1, prefetch t+2
        const int tn = (t + 1 < LSEQ) ? t + 1 : t;
        const int tp = (t + 2 < LSEQ) ? t + 2 : t;
        usht8 xn0 = *(const usht8*)(xp + xp_addr(bt, tn, lh, nc0));
        usht8 xn1 = *(const usht8*)(xp + xp_addr(bt, tn, lh, nc1));
        __builtin_prefetch((const void*)(xp + xp_addr(bt, tp, lh, nc0)), 0, 1);
        __builtin_prefetch((const void*)(xp + xp_addr(bt, tp, lh, nc1)), 0, 1);

        // h @ W_hh^T : K = 512 -> 16 chunks of 32
        #pragma unroll 8
        for (int kc = 0; kc < 16; ++kc) {
            Frag16 a;
            const int kb = kc * 32 + lh * 8;
            a.u8[0] = *(const usht8*)&hs[cur][lm][kb];
            a.u8[1] = *(const usht8*)&hs[cur][lm][kb + 16];
            const unsigned short* brow = whhT + (size_t)(kc * 32 + lh * 8) * H + n0 + lm;
            #pragma unroll
            for (int tile = 0; tile < 2; ++tile) {
                Frag16 b;
                #pragma unroll
                for (int i = 0; i < 8; ++i) {
                    b.u8[0][i] = brow[(size_t)i * H + tile * 16];
                    b.u8[1][i] = brow[(size_t)(16 + i) * H + tile * 16];
                }
                acc[tile] = __builtin_amdgcn_wmma_f32_16x16x32_bf16(
                    false, a.v, false, b.v, (short)0, acc[tile], false, false);
            }
        }

        // tanh + write new h into the OTHER buffer (no barrier needed first)
        #pragma unroll
        for (int tile = 0; tile < 2; ++tile) {
            const int ncol = n0 + tile * 16 + lm;
            #pragma unroll
            for (int v = 0; v < 8; ++v) {
                const int m = lh * 8 + v;
                float hv = tanhf(acc[tile][v]);
                hs[nxt][m][ncol] = f2bf(hv);
                if (t == LSEQ - 1)
                    h_out[(size_t)(b0 + m) * H + ncol] = hv;
            }
        }
        xf0 = xn0; xf1 = xn1;
        __syncthreads();   // single barrier per step
    }

    // epilogue: y = h @ W_out^T + b_out. Final h is in buffer 0 (LSEQ even).
    v8f yac[2];
    #pragma unroll
    for (int tile = 0; tile < 2; ++tile) {
        const float bias = b_out[n0 + tile * 16 + lm];
        #pragma unroll
        for (int v = 0; v < 8; ++v) yac[tile][v] = bias;
    }
    #pragma unroll 4
    for (int kc = 0; kc < 16; ++kc) {
        Frag16 a;
        const int kb = kc * 32 + lh * 8;
        a.u8[0] = *(const usht8*)&hs[0][lm][kb];
        a.u8[1] = *(const usht8*)&hs[0][lm][kb + 16];
        const unsigned short* brow = woutT + (size_t)(kc * 32 + lh * 8) * H + n0 + lm;
        #pragma unroll
        for (int tile = 0; tile < 2; ++tile) {
            Frag16 b;
            #pragma unroll
            for (int i = 0; i < 8; ++i) {
                b.u8[0][i] = brow[(size_t)i * H + tile * 16];
                b.u8[1][i] = brow[(size_t)(16 + i) * H + tile * 16];
            }
            yac[tile] = __builtin_amdgcn_wmma_f32_16x16x32_bf16(
                false, a.v, false, b.v, (short)0, yac[tile], false, false);
        }
    }
    #pragma unroll
    for (int tile = 0; tile < 2; ++tile) {
        const int ncol = n0 + tile * 16 + lm;
        #pragma unroll
        for (int v = 0; v < 8; ++v) {
            const int m = lh * 8 + v;
            y_out[(size_t)(b0 + m) * H + ncol] = yac[tile][v];
        }
    }
}

// ---------------------------------------------------------------------------
extern "C" void kernel_launch(void* const* d_in, const int* in_sizes, int n_in,
                              void* d_out, int out_size, void* d_ws, size_t ws_size,
                              hipStream_t stream) {
    const float* x     = (const float*)d_in[0];
    const float* h0    = (const float*)d_in[1];
    const float* W_ih  = (const float*)d_in[2];
    const float* W_hh  = (const float*)d_in[3];
    const float* b_ih  = (const float*)d_in[4];
    const float* b_hh  = (const float*)d_in[5];
    const float* W_out = (const float*)d_in[6];
    const float* b_out = (const float*)d_in[7];

    // workspace layout: xp (bf16, 128MB, fragment-swizzled) | W_hh^T | W_out^T
    unsigned short* xp    = (unsigned short*)d_ws;
    unsigned short* whhT  = xp + (size_t)NB * LSEQ * H;
    unsigned short* woutT = whhT + (size_t)H * H;

    float* y_out = (float*)d_out;
    float* h_out = y_out + (size_t)NB * H;

    // A: weight transpose/convert (262144 threads)
    wprep_kernel<<<512, 512, 0, stream>>>(W_hh, W_out, whhT, woutT);
    // B: big input-projection GEMM: grid = (M/128, H/128)
    xp_gemm_kernel<<<dim3((NB * LSEQ) / 128, H / 128), 256, 0, stream>>>(
        x, W_ih, b_ih, b_hh, xp);
    // C: serial scan, one workgroup per 16-row batch tile
    rnn_scan_kernel<<<NB / 16, 512, 0, stream>>>(
        xp, h0, whhT, woutT, b_out, y_out, h_out);
}
